// GaussianFeatureLeverage_5669356831062
// MI455X (gfx1250) — matrix operand order repaired and checked by reading, not compiled, not run on hardware
//
#include <hip/hip_runtime.h>
#include <math.h>

// ---------------- constants (match reference) ----------------
#define WB      2           // batch
#define H2      48
#define W2      160
#define NPIX    (H2*W2)     // 7680
#define CFEAT   2023
#define NBINS   96
#define COST_CH 1927        // cost channels start here inside feature
#define MIND    0.1f
#define MAXD    100.0f

typedef __bf16 bf16;
typedef __attribute__((ext_vector_type(16))) bf16  v16bf;
typedef __attribute__((ext_vector_type(8)))  float v8f;

static __device__ __forceinline__ float dev_sigmoid(float x){ return 1.f/(1.f+expf(-x)); }

// ---------------- ws layout (float offsets) ----------------
#define OFS_FEATURE 0ull
#define SZ_FEATURE  (2ull*CFEAT*NPIX)                 // 31,073,280
#define OFS_TMP     (OFS_FEATURE + SZ_FEATURE)
#define SZ_TMP      (2ull*CFEAT*NPIX)
#define OFS_MISS    (OFS_TMP + SZ_TMP)
#define SZ_MISS     (2ull*NBINS*NPIX)
#define OFS_DEPTH   (OFS_MISS + SZ_MISS)
#define SZ_DEPTH    (2ull*NPIX)
#define OFS_MEAN    (OFS_DEPTH + SZ_DEPTH)
#define SZ_MEAN     4096ull
#define OFS_HID     (OFS_MEAN + SZ_MEAN)
#define SZ_HID      512ull
#define OFS_SCALE   (OFS_HID + SZ_HID)
#define SZ_SCALE    4096ull
#define OFS_G1      (OFS_SCALE + SZ_SCALE)
#define SZ_G1       (2ull*22*NPIX)
#define OFS_GHEAD   (OFS_G1 + SZ_G1)
#define SZ_GHEAD    (2ull*11*NPIX)
#define OFS_POS     (OFS_GHEAD + SZ_GHEAD)
#define SZ_POS      (2ull*NPIX*3)
#define OFS_OPAC    (OFS_POS + SZ_POS)
#define SZ_OPAC     (2ull*NPIX)
#define OFS_GF      (OFS_OPAC + SZ_OPAC)
#define SZ_GF       (2ull*64*NPIX)
#define OFS_FACC    (OFS_GF + SZ_GF)
#define SZ_FACC     (2ull*NPIX*64)
#define OFS_WACC    (OFS_FACC + SZ_FACC)
#define SZ_WACC     (2ull*NPIX)
#define OFS_FUSED   (OFS_WACC + SZ_WACC)
#define SZ_FUSED    (2ull*228*NPIX)
#define OFS_WPACK   (OFS_FUSED + SZ_FUSED)
// largest packed weight: 9 * 2023 * 2048 bf16 -> 18.6M floats

// ---------------- bilinear resize into a channel slice of feature ----------------
__global__ void k_resize_into(const float* __restrict__ src, float* __restrict__ dst,
                              int cin, int hin, int win, int cofs) {
  int idx = blockIdx.x*blockDim.x + threadIdx.x;
  int total = WB*cin*NPIX;
  if (idx >= total) return;
  int p = idx % NPIX; int c = (idx/NPIX) % cin; int b = idx/(NPIX*cin);
  int x = p % W2, y = p / W2;
  float sy = (y + 0.5f) * ((float)hin/(float)H2) - 0.5f;
  float sx = (x + 0.5f) * ((float)win/(float)W2) - 0.5f;
  int y0 = (int)floorf(sy), x0 = (int)floorf(sx);
  float dy = sy - (float)y0, dx = sx - (float)x0;
  int y0c = min(max(y0,0),hin-1), y1c = min(max(y0+1,0),hin-1);
  int x0c = min(max(x0,0),win-1), x1c = min(max(x0+1,0),win-1);
  const float* s = src + ((size_t)b*cin + c)*(size_t)(hin*win);
  float v = (1.f-dy)*((1.f-dx)*s[y0c*win+x0c] + dx*s[y0c*win+x1c])
          +      dy *((1.f-dx)*s[y1c*win+x0c] + dx*s[y1c*win+x1c]);
  dst[((size_t)b*CFEAT + cofs + c)*NPIX + p] = v;
}

// ---------------- disp -> depth ----------------
__global__ void k_depth(const float* __restrict__ disp, float* __restrict__ depth) {
  int idx = blockIdx.x*blockDim.x + threadIdx.x;
  if (idx >= WB*NPIX) return;
  float mind = 1.f/MAXD, maxd = 1.f/MIND;
  float d = 1.f/(mind + (maxd-mind)*disp[idx]);
  depth[idx] = fminf(fmaxf(d, MIND), MAXD);
}

// ---------------- plane-sweep cost volume ----------------
__global__ void k_cost(const float* __restrict__ f0, const float* __restrict__ lf0,
                       const float* __restrict__ K2, const float* __restrict__ invK2,
                       const float* __restrict__ T,
                       float* __restrict__ feature, float* __restrict__ missing) {
  int idx = blockIdx.x*blockDim.x + threadIdx.x;
  int total = WB*NBINS*NPIX;
  if (idx >= total) return;
  int p = idx % NPIX; int d = (idx/NPIX) % NBINS; int b = idx/(NPIX*NBINS);
  float xf = (float)(p % W2), yf = (float)(p / W2);
  const float* iK = invK2 + b*16; const float* Km = K2 + b*16; const float* Tm = T + b*16;
  float cam0 = iK[0]*xf + iK[1]*yf + iK[2];
  float cam1 = iK[4]*xf + iK[5]*yf + iK[6];
  float cam2 = iK[8]*xf + iK[9]*yf + iK[10];
  float bin = MIND + (float)d*((MAXD-MIND)/(float)(NBINS-1));
  float P0 = bin*cam0, P1 = bin*cam1, P2 = bin*cam2;
  float w0 = Tm[0]*P0+Tm[1]*P1+Tm[2]*P2+Tm[3];
  float w1 = Tm[4]*P0+Tm[5]*P1+Tm[6]*P2+Tm[7];
  float w2 = Tm[8]*P0+Tm[9]*P1+Tm[10]*P2+Tm[11];
  float w3 = Tm[12]*P0+Tm[13]*P1+Tm[14]*P2+Tm[15];
  float pr0 = Km[0]*w0+Km[1]*w1+Km[2]*w2+Km[3]*w3;
  float pr1 = Km[4]*w0+Km[5]*w1+Km[6]*w2+Km[7]*w3;
  float pr2 = Km[8]*w0+Km[9]*w1+Km[10]*w2+Km[11]*w3;
  float z  = fmaxf(pr2, 1e-3f);
  float px = pr0/z, py = pr1/z;
  float x0 = floorf(px), y0 = floorf(py);
  float ddx = px-x0, ddy = py-y0;
  int xi0 = (int)x0, yi0 = (int)y0;
  float wgt[4] = {(1.f-ddx)*(1.f-ddy), ddx*(1.f-ddy), (1.f-ddx)*ddy, ddx*ddy};
  int xs[4] = {xi0, xi0+1, xi0, xi0+1};
  int ys[4] = {yi0, yi0, yi0+1, yi0+1};
  const float* fb = f0  + (size_t)b*64*NPIX;
  const float* lb = lf0 + (size_t)b*64*NPIX;
  float acc = 0.f;
  for (int c = 0; c < 64; c++) {
    float warped = 0.f;
    #pragma unroll
    for (int k = 0; k < 4; k++) {
      int xi = xs[k], yi = ys[k];
      bool ok = (xi>=0) && (xi<W2) && (yi>=0) && (yi<H2);
      int xig = min(max(xi,0),W2-1), yig = min(max(yi,0),H2-1);
      warped += ok ? lb[(size_t)c*NPIX + yig*W2 + xig]*wgt[k] : 0.f;
    }
    acc += fabsf(warped - fb[(size_t)c*NPIX + p]);
  }
  float diff = acc * (1.f/64.f);
  float inb = (px>=0.f && px<=(float)(W2-1) && py>=0.f && py<=(float)(H2-1)) ? 1.f : 0.f;
  feature[((size_t)b*CFEAT + COST_CH + d)*NPIX + p] = diff*inb;
  missing[((size_t)b*NBINS + d)*NPIX + p] = 1.f - inb;
}

__global__ void k_conf(float* __restrict__ feature, const float* __restrict__ missing) {
  int idx = blockIdx.x*blockDim.x + threadIdx.x;
  if (idx >= WB*NPIX) return;
  int p = idx % NPIX, b = idx / NPIX;
  int cnt = 0;
  for (int d = 0; d < NBINS; d++) {
    float c = feature[((size_t)b*CFEAT + COST_CH + d)*NPIX + p];
    float m = missing[((size_t)b*NBINS + d)*NPIX + p];
    if (c*(1.f-m) > 0.f) cnt++;
  }
  float conf = (cnt == NBINS) ? 1.f : 0.f;
  for (int d = 0; d < NBINS; d++)
    feature[((size_t)b*CFEAT + COST_CH + d)*NPIX + p] *= conf;
}

// ---------------- per-channel spatial mean (LDS reduction) ----------------
__global__ __launch_bounds__(256) void k_cmean(const float* __restrict__ x,
                                               float* __restrict__ mean, int HW) {
  int bc = blockIdx.x;
  __shared__ float sm[256];
  const float* xp = x + (size_t)bc*HW;
  float s = 0.f;
  for (int i = threadIdx.x; i < HW; i += 256) s += xp[i];
  sm[threadIdx.x] = s; __syncthreads();
  for (int st = 128; st > 0; st >>= 1) {
    if (threadIdx.x < st) sm[threadIdx.x] += sm[threadIdx.x + st];
    __syncthreads();
  }
  if (threadIdx.x == 0) mean[bc] = sm[0] / (float)HW;
}

// ---------------- small dense layer: y = act(W x + b), W [Cout][Cin] ----------------
__global__ void k_dense(const float* __restrict__ x, const float* __restrict__ W,
                        const float* __restrict__ bias, float* __restrict__ y,
                        int Cin, int Cout, int act /*0 relu, 1 sigmoid*/) {
  int idx = blockIdx.x*blockDim.x + threadIdx.x;
  if (idx >= WB*Cout) return;
  int j = idx % Cout, b = idx / Cout;
  const float* xb = x + (size_t)b*Cin;
  const float* wr = W + (size_t)j*Cin;
  float s = bias[j];
  for (int c = 0; c < Cin; c++) s += xb[c]*wr[c];
  y[idx] = (act == 0) ? fmaxf(s, 0.f) : dev_sigmoid(s);
}

// ---------------- pack conv weights to bf16 [tap][Cout][Kpad] (zero-filled tail) ----
__global__ void k_pack_w(const float* __restrict__ w, bf16* __restrict__ wp,
                         int Cin, int Kpad, int Cout) {
  int idx = blockIdx.x*blockDim.x + threadIdx.x;
  int total = 9*Cout*Kpad;
  if (idx >= total) return;
  int c  = idx % Kpad;
  int co = (idx / Kpad) % Cout;
  int t  = idx / (Kpad*Cout);
  float v = (c < Cin) ? w[((size_t)co*Cin + c)*9 + t] : 0.f;
  wp[idx] = (bf16)v;
}

// ---------------- implicit-GEMM conv3x3 via bf16 WMMA ----------------
// Tap-decomposed (9 GEMMs, K=Cin). 4 waves/block share one 16-pixel tile.
// 64ch x 16pix bf16 B-panels are staged in ping-pong LDS buffers (1 barrier
// per 2 WMMAs); A comes from pre-packed bf16 weights (contiguous 16B loads).
// Next K-chunk is pulled with global_prefetch while the WMMAs execute.
__global__ __launch_bounds__(128) void k_conv_wmma(
    const float* __restrict__ in, const float* __restrict__ sescale,
    const bf16* __restrict__ wpak, const float* __restrict__ bias,
    float* __restrict__ out,
    int Cin, int Kpad, int Cout, int H, int W, int padEdge,
    int act /*0 none,1 ELU,2 GELU*/) {
  __shared__ __attribute__((aligned(16))) bf16 ldsB[2][16][72]; // [buf][pixel][ch(+pad)]
  __shared__ int ysh[16*3];
  __shared__ int xsh[16*3];

  const int tid  = threadIdx.x;
  const int lane = tid & 31;
  const int wv   = tid >> 5;
  const int hi   = lane >> 4;
  const int lm   = lane & 15;
  const int ifix = tid & 15;     // pixel handled in fills (constant per thread)
  const int cgrp = tid >> 4;     // channel subgroup 0..7 in fills
  const int HWp  = H*W;
  const int b    = blockIdx.z;
  const int pix0 = blockIdx.x*16;
  const int co0  = (blockIdx.y*4 + wv)*16;
  const int p    = pix0 + lm;
  const int arow = co0 + lm;
  const int arowc = (arow < Cout) ? arow : 0;
  const float* inb = in + (size_t)b*Cin*HWp;
  const float* sc  = sescale ? (sescale + (size_t)b*Cin) : (const float*)0;

  // precompute padded neighbor coords once per tile (16 pixels x 3 offsets)
  if (tid < 16) {
    int pp = pix0 + tid; if (pp >= HWp) pp = HWp - 1;
    int y = pp / W, x = pp % W;
    #pragma unroll
    for (int d = 0; d < 3; d++) {
      int yy = y + d - 1, xx = x + d - 1;
      if (padEdge) { yy = min(max(yy,0),H-1); xx = min(max(xx,0),W-1); }
      else {
        yy = (yy < 0) ? -yy : yy; yy = (yy > H-1) ? (2*H-2-yy) : yy;
        xx = (xx < 0) ? -xx : xx; xx = (xx > W-1) ? (2*W-2-xx) : xx;
      }
      ysh[tid*3+d] = yy; xsh[tid*3+d] = xx;
    }
  }
  __syncthreads();

  v8f acc = {0.f,0.f,0.f,0.f,0.f,0.f,0.f,0.f};
  int buf = 0;
  #pragma unroll
  for (int t = 0; t < 9; t++) {
    const int dyi = t/3, dxi = t - (t/3)*3;
    // per-tap, per-thread row offset for fills (hoisted out of K loop)
    const size_t roff = (size_t)(ysh[ifix*3 + dyi]*W + xsh[ifix*3 + dxi]);
    const bf16* wrow = wpak + ((size_t)t*Cout + arowc)*Kpad;
    for (int kk = 0; kk < Kpad; kk += 64) {
      // ---- cooperative fill of 64ch x 16pix panel (8 elements/thread) ----
      #pragma unroll
      for (int e = 0; e < 8; e++) {
        int c  = cgrp + 8*e;          // 0..63
        int cg = kk + c;
        float v = 0.f;
        if (cg < Cin) {
          v = inb[(size_t)cg*HWp + roff];
          if (sc) v *= sc[cg];
        }
        ldsB[buf][ifix][c] = (bf16)v;
      }
      // speculative prefetch of next chunk (global_prefetch path)
      if (kk + 64 < Kpad) {
        int cg2 = kk + 64 + cgrp;
        if (cg2 < Cin) __builtin_prefetch(&inb[(size_t)cg2*HWp + roff], 0, 3);
      }
      __syncthreads();   // ping-pong: single barrier per panel
      // ---- consume: two K=32 WMMAs ----
      v16bf a0, a1, b0, b1;
      __builtin_memcpy(&a0,            wrow + kk      + hi*8, 16);
      __builtin_memcpy((char*)&a0 + 16, wrow + kk + 16 + hi*8, 16);
      __builtin_memcpy(&a1,            wrow + kk + 32 + hi*8, 16);
      __builtin_memcpy((char*)&a1 + 16, wrow + kk + 48 + hi*8, 16);
      __builtin_memcpy(&b0,            &ldsB[buf][lm][hi*8],      16);
      __builtin_memcpy((char*)&b0 + 16, &ldsB[buf][lm][16 + hi*8], 16);
      __builtin_memcpy(&b1,            &ldsB[buf][lm][32 + hi*8], 16);
      __builtin_memcpy((char*)&b1 + 16, &ldsB[buf][lm][48 + hi*8], 16);
      acc = __builtin_amdgcn_wmma_f32_16x16x32_bf16(false, a0, false, b0,
                                                    (short)0, acc, false, false);
      acc = __builtin_amdgcn_wmma_f32_16x16x32_bf16(false, a1, false, b1,
                                                    (short)0, acc, false, false);
      buf ^= 1;
    }
  }
  // D layout: VGPR v -> cout = co0 + v + 8*hi, pixel col = lane&15
  #pragma unroll
  for (int v8 = 0; v8 < 8; v8++) {
    int co = co0 + v8 + 8*hi;
    if (co < Cout && p < HWp) {
      float v = acc[v8] + bias[co];
      if (act == 1)       v = (v > 0.f) ? v : (expf(v) - 1.f);
      else if (act == 2)  { float tt = v;
        v = 0.5f*tt*(1.f + tanhf(0.79788456f*(tt + 0.044715f*tt*tt*tt))); }
      out[((size_t)b*Cout + co)*HWp + p] = v;
    }
  }
}

// ---------------- backproject + offsets -> positions / opacity ----------------
__global__ void k_positions(const float* __restrict__ ghead, const float* __restrict__ depth,
                            const float* __restrict__ invK2,
                            float* __restrict__ pos, float* __restrict__ opac) {
  int idx = blockIdx.x*blockDim.x + threadIdx.x;
  if (idx >= WB*NPIX) return;
  int p = idx % NPIX, b = idx / NPIX;
  float xf = (float)(p % W2), yf = (float)(p / W2);
  const float* iK = invK2 + b*16;
  float cam0 = iK[0]*xf + iK[1]*yf + iK[2];
  float cam1 = iK[4]*xf + iK[5]*yf + iK[6];
  float cam2 = iK[8]*xf + iK[9]*yf + iK[10];
  float dep = depth[idx];
  const float* g = ghead + (size_t)b*11*NPIX;
  pos[((size_t)idx)*3 + 0] = dep*cam0 + g[0*NPIX + p];
  pos[((size_t)idx)*3 + 1] = dep*cam1 + g[1*NPIX + p];
  pos[((size_t)idx)*3 + 2] = dep*cam2 + g[2*NPIX + p];
  opac[idx] = dev_sigmoid(g[3*NPIX + p]);
}

__global__ void k_zero(float* __restrict__ p, size_t n) {
  size_t i = (size_t)blockIdx.x*blockDim.x + threadIdx.x;
  if (i < n) p[i] = 0.f;
}

// ---------------- point-splat rasterize (atomic scatter) ----------------
__global__ void k_raster(const float* __restrict__ pos, const float* __restrict__ opac,
                         const float* __restrict__ gf, const float* __restrict__ Kmat,
                         float* __restrict__ facc, float* __restrict__ wacc, int h, int w) {
  int idx = blockIdx.x*blockDim.x + threadIdx.x;
  if (idx >= WB*NPIX) return;
  int n = idx % NPIX, b = idx / NPIX;
  const float* Km = Kmat + b*16;
  float P0 = pos[(size_t)idx*3+0], P1 = pos[(size_t)idx*3+1], P2 = pos[(size_t)idx*3+2];
  float pr0 = Km[0]*P0 + Km[1]*P1 + Km[2]*P2;
  float pr1 = Km[4]*P0 + Km[5]*P1 + Km[6]*P2;
  float pr2 = Km[8]*P0 + Km[9]*P1 + Km[10]*P2;
  float z  = pr2, zc = fmaxf(pr2, 1e-3f);
  float sx = pr0/zc, sy = pr1/zc;
  float valid = (z > MIND && z < MAXD) ? 1.f : 0.f;
  float op = opac[idx]*valid;
  float x0 = floorf(sx), y0 = floorf(sy);
  float dx = sx - x0, dy = sy - y0;
  const float* gfb = gf + (size_t)b*64*NPIX;
  #pragma unroll
  for (int k = 0; k < 4; k++) {
    int ox = k & 1, oy = k >> 1;
    float wgt = (ox ? dx : 1.f-dx) * (oy ? dy : 1.f-dy);
    float xi = x0 + (float)ox, yi = y0 + (float)oy;
    float inb = (xi >= 0.f && xi < (float)w && yi >= 0.f && yi < (float)h) ? 1.f : 0.f;
    int xig = min(max((int)xi, 0), w-1), yig = min(max((int)yi, 0), h-1);
    int id2 = b*h*w + yig*w + xig;
    float ww = wgt*op*inb;
    if (ww != 0.f) {
      atomicAdd(&wacc[id2], ww);
      for (int c = 0; c < 64; c++)
        atomicAdd(&facc[(size_t)id2*64 + c], ww*gfb[(size_t)c*NPIX + n]);
    }
  }
}

__global__ void k_fused_gs(const float* __restrict__ facc, const float* __restrict__ wacc,
                           float* __restrict__ fused, int Cf, int hw) {
  int idx = blockIdx.x*blockDim.x + threadIdx.x;
  if (idx >= WB*64*hw) return;
  int p = idx % hw, c = (idx/hw) % 64, b = idx/(hw*64);
  int id2 = b*hw + p;
  fused[((size_t)b*Cf + c)*hw + p] = facc[(size_t)id2*64 + c] / (wacc[id2] + 1e-8f);
}

__global__ void k_copy_ch(const float* __restrict__ src, float* __restrict__ dst,
                          int srcCtot, int srcCofs, int Cs, int Cf, int cofs, int hw) {
  int idx = blockIdx.x*blockDim.x + threadIdx.x;
  if (idx >= WB*Cs*hw) return;
  int p = idx % hw, c = (idx/hw) % Cs, b = idx/(hw*Cs);
  dst[((size_t)b*Cf + cofs + c)*hw + p] = src[((size_t)b*srcCtot + srcCofs + c)*hw + p];
}

// ---------------- host orchestration ----------------
static inline int cdiv(int a, int b){ return (a + b - 1)/b; }

static void launch_conv(const float* in, const float* sescale,
                        const float* w, const float* bias, float* out,
                        bf16* wpack, int Cin, int Cout, int H, int W,
                        int padEdge, int act, hipStream_t stream) {
  int Kpad = cdiv(Cin, 64)*64;
  int tot = 9*Cout*Kpad;
  k_pack_w<<<cdiv(tot,256),256,0,stream>>>(w, wpack, Cin, Kpad, Cout);
  dim3 g(cdiv(H*W,16), cdiv(Cout,64), WB);
  k_conv_wmma<<<g,128,0,stream>>>(in, sescale, wpack, bias, out,
                                  Cin, Kpad, Cout, H, W, padEdge, act);
}

extern "C" void kernel_launch(void* const* d_in, const int* in_sizes, int n_in,
                              void* d_out, int out_size, void* d_ws, size_t ws_size,
                              hipStream_t stream) {
  (void)in_sizes; (void)n_in; (void)out_size; (void)ws_size;
  float* ws = (float*)d_ws;
  float* feature = ws + OFS_FEATURE;
  float* tmp     = ws + OFS_TMP;
  float* miss    = ws + OFS_MISS;
  float* depth   = ws + OFS_DEPTH;
  float* meanb   = ws + OFS_MEAN;
  float* hidb    = ws + OFS_HID;
  float* scaleb  = ws + OFS_SCALE;
  float* g1      = ws + OFS_G1;
  float* ghead   = ws + OFS_GHEAD;
  float* pos     = ws + OFS_POS;
  float* opac    = ws + OFS_OPAC;
  float* gf      = ws + OFS_GF;
  float* facc    = ws + OFS_FACC;
  float* wacc    = ws + OFS_WACC;
  float* fused   = ws + OFS_FUSED;
  bf16*  wpack   = (bf16*)(ws + OFS_WPACK);
  float* outp    = (float*)d_out;

  // --- input pointer map (setup_inputs insertion order, params recursively) ---
  const float* feat[4] = {(const float*)d_in[0], (const float*)d_in[2],
                          (const float*)d_in[4], (const float*)d_in[6]};
  const float* lfp [4] = {(const float*)d_in[1], (const float*)d_in[3],
                          (const float*)d_in[5], (const float*)d_in[7]};
  const float* cols[4] = {(const float*)d_in[8],  (const float*)d_in[10],
                          (const float*)d_in[12], (const float*)d_in[14]};
  const float* dsps[4] = {(const float*)d_in[9],  (const float*)d_in[11],
                          (const float*)d_in[13], (const float*)d_in[15]};
  const float* K2 = (const float*)d_in[16];
  const float* invK2 = (const float*)d_in[17];
  const float* Ks[4] = {K2, (const float*)d_in[18], (const float*)d_in[19],
                        (const float*)d_in[20]};
  const float* lookupT = (const float*)d_in[21];
  const float* h_se_w1 = (const float*)d_in[22]; const float* h_se_b1 = (const float*)d_in[23];
  const float* h_se_w2 = (const float*)d_in[24]; const float* h_se_b2 = (const float*)d_in[25];
  const float* h_w1 = (const float*)d_in[26]; const float* h_b1 = (const float*)d_in[27];
  const float* h_w2 = (const float*)d_in[28]; const float* h_b2 = (const float*)d_in[29];
  const float* l_w1 = (const float*)d_in[30]; const float* l_b1 = (const float*)d_in[31];
  const float* l_w2 = (const float*)d_in[32]; const float* l_b2 = (const float*)d_in[33];

  const int TB = 256;
  // ---- build 2023-channel feature ----
  const int featC[4] = {64,128,256,512};
  const int hsz[4] = {48,24,12,6}, wsz[4] = {160,80,40,20};
  int cofs = 0;
  for (int i = 0; i < 4; i++) {
    int tot = WB*featC[i]*NPIX;
    k_resize_into<<<cdiv(tot,TB),TB,0,stream>>>(feat[i], feature, featC[i], hsz[i], wsz[i], cofs);
    cofs += featC[i];
    k_resize_into<<<cdiv(tot,TB),TB,0,stream>>>(lfp[i],  feature, featC[i], hsz[i], wsz[i], cofs);
    cofs += featC[i];
  }
  for (int i = 0; i < 4; i++) {
    k_resize_into<<<cdiv(WB*NPIX,TB),TB,0,stream>>>(dsps[i], feature, 1, hsz[i], wsz[i], cofs);
    cofs += 1;
  }
  k_resize_into<<<cdiv(WB*3*NPIX,TB),TB,0,stream>>>(cols[0], feature, 3, H2, W2, cofs);

  // ---- depth + cost volume ----
  k_depth<<<cdiv(WB*NPIX,TB),TB,0,stream>>>(dsps[0], depth);
  k_cost<<<cdiv(WB*NBINS*NPIX,TB),TB,0,stream>>>(feat[0], lfp[0], K2, invK2, lookupT,
                                                 feature, miss);
  k_conf<<<cdiv(WB*NPIX,TB),TB,0,stream>>>(feature, miss);

  // ---- head: SE + conv(GELU) + conv ----
  k_cmean<<<WB*CFEAT,256,0,stream>>>(feature, meanb, NPIX);
  k_dense<<<cdiv(WB*126,128),128,0,stream>>>(meanb, h_se_w1, h_se_b1, hidb, CFEAT, 126, 0);
  k_dense<<<cdiv(WB*CFEAT,128),128,0,stream>>>(hidb, h_se_w2, h_se_b2, scaleb, 126, CFEAT, 1);
  launch_conv(feature, scaleb, h_w1, h_b1, g1, wpack, CFEAT, 22, H2, W2,
              /*edge*/1, /*gelu*/2, stream);
  launch_conv(g1, (const float*)0, h_w2, h_b2, ghead, wpack, 22, 11, H2, W2,
              /*edge*/1, /*none*/0, stream);
  k_positions<<<cdiv(WB*NPIX,TB),TB,0,stream>>>(ghead, depth, invK2, pos, opac);

  // ---- leve: conv 2023->2023 (ELU) then 2023->64 (ELU) — the big WMMA GEMMs ----
  launch_conv(feature, (const float*)0, l_w1, l_b1, tmp, wpack, CFEAT, CFEAT, H2, W2,
              /*reflect*/0, /*elu*/1, stream);
  launch_conv(tmp, (const float*)0, l_w2, l_b2, gf, wpack, CFEAT, 64, H2, W2,
              /*reflect*/0, /*elu*/1, stream);

  // ---- per-scale rasterize + resume ----
  const int Ci[4]  = {228,196,324,580};
  const int Cri[4] = {14,12,20,36};
  const int Co[4]  = {18,36,72,144};
  const size_t outofs[4] = {0, 276480, 414720, 483840};
  for (int i = 0; i < 4; i++) {
    int h = hsz[i], w = wsz[i], hw = h*w;
    const float* rp_se_w1 = (const float*)d_in[34 + 6*i + 0];
    const float* rp_se_b1 = (const float*)d_in[34 + 6*i + 1];
    const float* rp_se_w2 = (const float*)d_in[34 + 6*i + 2];
    const float* rp_se_b2 = (const float*)d_in[34 + 6*i + 3];
    const float* rp_w     = (const float*)d_in[34 + 6*i + 4];
    const float* rp_b     = (const float*)d_in[34 + 6*i + 5];

    k_zero<<<cdiv(WB*hw*64,TB),TB,0,stream>>>(facc, (size_t)WB*hw*64);
    k_zero<<<cdiv(WB*hw,TB),TB,0,stream>>>(wacc, (size_t)WB*hw);
    k_raster<<<cdiv(WB*NPIX,TB),TB,0,stream>>>(pos, opac, gf, Ks[i], facc, wacc, h, w);

    // assemble fused = [gs_feat(64) | feat_i | color_i | disp_i | (cost if i==0)]
    k_fused_gs<<<cdiv(WB*64*hw,TB),TB,0,stream>>>(facc, wacc, fused, Ci[i], hw);
    int co2 = 64;
    k_copy_ch<<<cdiv(WB*featC[i]*hw,TB),TB,0,stream>>>(feat[i], fused, featC[i], 0,
                                                       featC[i], Ci[i], co2, hw);
    co2 += featC[i];
    k_copy_ch<<<cdiv(WB*3*hw,TB),TB,0,stream>>>(cols[i], fused, 3, 0, 3, Ci[i], co2, hw);
    co2 += 3;
    k_copy_ch<<<cdiv(WB*1*hw,TB),TB,0,stream>>>(dsps[i], fused, 1, 0, 1, Ci[i], co2, hw);
    co2 += 1;
    if (i == 0) {
      k_copy_ch<<<cdiv(WB*NBINS*hw,TB),TB,0,stream>>>(feature, fused, CFEAT, COST_CH,
                                                      NBINS, Ci[i], co2, hw);
    }
    // SE over fused
    k_cmean<<<WB*Ci[i],256,0,stream>>>(fused, meanb, hw);
    k_dense<<<cdiv(WB*Cri[i],128),128,0,stream>>>(meanb, rp_se_w1, rp_se_b1, hidb,
                                                  Ci[i], Cri[i], 0);
    k_dense<<<cdiv(WB*Ci[i],128),128,0,stream>>>(hidb, rp_se_w2, rp_se_b2, scaleb,
                                                 Cri[i], Ci[i], 1);
    // conv_block (reflect pad + ELU) -> output, SE scale folded into input load
    launch_conv(fused, scaleb, rp_w, rp_b, outp + outofs[i], wpack,
                Ci[i], Co[i], h, w, /*reflect*/0, /*elu*/1, stream);
  }
}
